// MultiHeadAttention_11793980195310
// MI455X (gfx1250) — matrix operand order repaired and checked by reading, not compiled, not run on hardware
//
#include <hip/hip_runtime.h>

// ---------------------------------------------------------------------------
// CDNA5 WMMA (wave32): v_wmma_f32_16x16x32_bf16
// ---------------------------------------------------------------------------
typedef __attribute__((ext_vector_type(16))) __bf16 v16bf;
typedef __attribute__((ext_vector_type(8)))  float  v8f;
typedef int v4i __attribute__((vector_size(16)));   // builtin param pointee type

#if __has_builtin(__builtin_amdgcn_global_load_async_to_lds_b128)
#define HAVE_ASYNC_LDS 1
#else
#define HAVE_ASYNC_LDS 0
#endif

static __device__ __forceinline__ __bf16 bits_to_bf16(unsigned short u) {
    return __builtin_bit_cast(__bf16, u);
}

static __device__ __forceinline__ unsigned short f32_to_bf16_bits(float f) {
    unsigned u = __builtin_bit_cast(unsigned, f);
    unsigned r = u + 0x7fffu + ((u >> 16) & 1u);   // round-to-nearest-even
    return (unsigned short)(r >> 16);
}

static __device__ __forceinline__ v8f wmma_bf16(v16bf a, v16bf b, v8f c) {
    return __builtin_amdgcn_wmma_f32_16x16x32_bf16(false, a, false, b,
                                                   (short)0, c, false, false);
}

// 16B global -> LDS copy via GLOBAL_LOAD_ASYNC_TO_LDS_B128 (ASYNCcnt).
static __device__ __forceinline__ void copy16_g2l(const unsigned short* g,
                                                  unsigned short* l) {
#if HAVE_ASYNC_LDS
    __builtin_amdgcn_global_load_async_to_lds_b128(
        (__attribute__((address_space(1))) v4i*)(unsigned long long)g,
        (__attribute__((address_space(3))) v4i*)(unsigned)(unsigned long long)l,
        0, 0);
#else
    *(uint4*)l = *(const uint4*)g;
#endif
}

// Async ops complete in order per wave: waiting <=N with 2N outstanding
// guarantees the older batch has landed while the next stays in flight.
static __device__ __forceinline__ void async_wait_le8() {
#if HAVE_ASYNC_LDS
    asm volatile("s_wait_asynccnt 0x8" ::: "memory");
#endif
}
static __device__ __forceinline__ void async_wait_le4() {
#if HAVE_ASYNC_LDS
    asm volatile("s_wait_asynccnt 0x4" ::: "memory");
#endif
}
static __device__ __forceinline__ void async_wait_le0() {
#if HAVE_ASYNC_LDS
    asm volatile("s_wait_asynccnt 0x0" ::: "memory");
#endif
}

// Split workgroup barrier that publishes LDS (dscnt) but does NOT drain
// loadcnt/asynccnt like __syncthreads' fence would — keeps async copies
// and global loads in flight across the barrier.
static __device__ __forceinline__ void wg_barrier() {
    asm volatile("s_wait_dscnt 0x0\n"
                 "s_barrier_signal -1\n"
                 "s_barrier_wait -1" ::: "memory");
}

// A fragment (16 rows x 32 K), contiguous K order (k = 16h + s) -> 32B contiguous / lane
static __device__ __forceinline__ v16bf load_a_frag(const unsigned short* base,
                                                    int stride, int lane) {
    int row   = lane & 15;
    int khalf = (lane & 16) ? 16 : 0;
    const unsigned short* p = base + (size_t)row * stride + khalf;
    v16bf a;
#pragma unroll
    for (int i = 0; i < 8; ++i) a[i] = bits_to_bf16(p[i]);
#pragma unroll
    for (int i = 0; i < 8; ++i) a[8 + i] = bits_to_bf16(p[8 + i]);
    return a;
}

// B fragment (32 K x 16 N) from K-contiguous tile -> 32B contiguous / lane
static __device__ __forceinline__ v16bf load_b_frag(const unsigned short* base,
                                                    int stride, int lane) {
    int n  = lane & 15;
    int kb = (lane & 16) ? 16 : 0;
    const unsigned short* p = base + (size_t)n * stride + kb;
    v16bf b;
#pragma unroll
    for (int i = 0; i < 16; ++i) b[i] = bits_to_bf16(p[i]);
    return b;
}

// ---------------------------------------------------------------------------
// fp32 -> bf16 convert (4-wide) for x
// ---------------------------------------------------------------------------
__global__ void k_f32_to_bf16(const float4* __restrict__ src,
                              uint2* __restrict__ dst, int n4) {
    int i = blockIdx.x * blockDim.x + threadIdx.x;
    if (i < n4) {
        float4 f = src[i];
        uint2 o;
        o.x = (unsigned)f32_to_bf16_bits(f.x) | ((unsigned)f32_to_bf16_bits(f.y) << 16);
        o.y = (unsigned)f32_to_bf16_bits(f.z) | ((unsigned)f32_to_bf16_bits(f.w) << 16);
        dst[i] = o;
    }
}

// ---------------------------------------------------------------------------
// fp32 -> bf16 + transpose for weights: dst[n][k] = (bf16) src[k][n].
// Pre-transposing makes every GEMM staging path a pure row-major async
// b128 copy (no ds transpose stores in hot loops).
// ---------------------------------------------------------------------------
__global__ __launch_bounds__(256)
void k_f32_to_bf16_T(const float* __restrict__ src,
                     unsigned short* __restrict__ dst) {
    __shared__ unsigned short T[32][33];
    const int k0 = blockIdx.y * 32, n0 = blockIdx.x * 32;
    const int tx = threadIdx.x & 31, ty = threadIdx.x >> 5;   // 32 x 8
#pragma unroll
    for (int it = 0; it < 4; ++it)
        T[ty + it * 8][tx] =
            f32_to_bf16_bits(src[(size_t)(k0 + ty + it * 8) * 1024 + n0 + tx]);
    __syncthreads();
#pragma unroll
    for (int it = 0; it < 4; ++it)
        dst[(size_t)(n0 + ty + it * 8) * 1024 + k0 + tx] = T[tx][ty + it * 8];
}

// ---------------------------------------------------------------------------
// Double-buffered bf16 GEMM mainloop: C tile [128x128] of A[8192x1024] @
// BT^T (BT = pre-transposed weight [n][k]).  K-step 64, 16 WMMA per step
// per wave; both tiles staged via async global->LDS b128 with the next
// tile's 8 copies kept in flight (s_wait_asynccnt 8).
// ---------------------------------------------------------------------------
static __device__ __forceinline__ void gemm_mainloop(
    const unsigned short* __restrict__ A, const unsigned short* __restrict__ BT,
    int M0, int N0, int tid, int lane, int wy, int wx,
    unsigned short (*As)[128][72], unsigned short (*Bs)[128][72],
    v8f acc[4][2]) {
    const int r = tid >> 3, c = (tid & 7) * 8;   // 32 rows/pass x 64 k, 4 passes

#pragma unroll
    for (int it = 0; it < 4; ++it) {             // prologue: tile 0
        copy16_g2l(A  + (size_t)(M0 + r + it * 32) * 1024 + c, &As[0][r + it * 32][c]);
        copy16_g2l(BT + (size_t)(N0 + r + it * 32) * 1024 + c, &Bs[0][r + it * 32][c]);
    }

    for (int t = 0; t < 16; ++t) {
        const int cur = t & 1, nxt = cur ^ 1;
        const int k1  = (t + 1) * 64;
        if (t < 15) {
#pragma unroll
            for (int it = 0; it < 4; ++it) {
                copy16_g2l(A  + (size_t)(M0 + r + it * 32) * 1024 + k1 + c,
                           &As[nxt][r + it * 32][c]);
                copy16_g2l(BT + (size_t)(N0 + r + it * 32) * 1024 + k1 + c,
                           &Bs[nxt][r + it * 32][c]);
            }
            if (t < 14)                       // prime L2 for tile t+2
                __builtin_prefetch(A + (size_t)(M0 + r) * 1024 + k1 + 64 + c, 0, 3);
            async_wait_le8();                 // tile t landed, t+1 in flight
        } else {
            async_wait_le0();
        }
        wg_barrier();

#pragma unroll
        for (int ks = 0; ks < 2; ++ks) {
            v16bf bf0 = load_b_frag(&Bs[cur][wx * 32][ks * 32], 72, lane);
            v16bf bf1 = load_b_frag(&Bs[cur][wx * 32 + 16][ks * 32], 72, lane);
#pragma unroll
            for (int i = 0; i < 4; ++i) {
                v16bf af = load_a_frag(&As[cur][wy * 64 + i * 16][ks * 32], 72, lane);
                acc[i][0] = wmma_bf16(af, bf0, acc[i][0]);
                acc[i][1] = wmma_bf16(af, bf1, acc[i][1]);
            }
        }
        wg_barrier();   // all waves done reading buf `cur` before refill
    }
}

// ---------------------------------------------------------------------------
// Fused QKV GEMM: blockIdx.z selects {WqT,WkT,WvT}.  Q,K row-major bf16
// [8192][1024]; V stored transposed: VT[b][h*64+d][n].
// ---------------------------------------------------------------------------
__global__ __launch_bounds__(256)
void k_qkv_gemm(const unsigned short* __restrict__ xb,
                const unsigned short* __restrict__ WqT,
                const unsigned short* __restrict__ WkT,
                const unsigned short* __restrict__ WvT,
                unsigned short* __restrict__ Qo,
                unsigned short* __restrict__ Ko,
                unsigned short* __restrict__ VTo) {
    const int which = blockIdx.z;
    const unsigned short* W = (which == 0) ? WqT : (which == 1) ? WkT : WvT;

    const int M0 = blockIdx.y * 128;
    const int N0 = blockIdx.x * 128;

    __shared__ __align__(16) unsigned short As[2][128][72];
    __shared__ __align__(16) unsigned short Bs[2][128][72];

    const int tid  = threadIdx.x;
    const int lane = tid & 31;
    const int wave = tid >> 5;
    const int wy   = wave >> 2;
    const int wx   = wave & 3;

    v8f acc[4][2] = {};
    gemm_mainloop(xb, W, M0, N0, tid, lane, wy, wx, As, Bs, acc);

    const int ncol = lane & 15;
    const int mofs = (lane & 16) ? 8 : 0;
    if (which != 2) {                        // uniform branch, hoisted
        unsigned short* Out = (which == 0) ? Qo : Ko;
#pragma unroll
        for (int i = 0; i < 4; ++i) {
            unsigned short* prow =
                Out + (size_t)(M0 + wy * 64 + i * 16 + mofs) * 1024;
#pragma unroll
            for (int j = 0; j < 2; ++j) {
                int col = N0 + wx * 32 + j * 16 + ncol;
#pragma unroll
                for (int r = 0; r < 8; ++r)
                    prow[(size_t)r * 1024 + col] = f32_to_bf16_bits(acc[i][j][r]);
            }
        }
    } else {                                 // V: transposed, contiguous rows
        const int bb  = M0 >> 11;            // batch (128-row tile within one b)
        const int nn0 = (M0 & 2047) + wy * 64 + mofs;
#pragma unroll
        for (int j = 0; j < 2; ++j) {
            int col = N0 + wx * 32 + j * 16 + ncol;
            unsigned short* pcol = VTo + ((size_t)bb * 1024 + col) * 2048 + nn0;
#pragma unroll
            for (int i = 0; i < 4; ++i)
#pragma unroll
                for (int r = 0; r < 8; ++r)
                    pcol[i * 16 + r] = f32_to_bf16_bits(acc[i][j][r]);
        }
    }
}

// ---------------------------------------------------------------------------
// Causal flash attention.  Grid (32 q-blocks, 64 b*h), 128 threads (4 waves,
// 16 query rows each).  K/V double-buffered: next 32-key block's 4 async
// copies stay in flight (s_wait_asynccnt 4) across softmax + WMMA.
// ---------------------------------------------------------------------------
__global__ __launch_bounds__(128)
void k_flash_attn(const unsigned short* __restrict__ Q,
                  const unsigned short* __restrict__ K,
                  const unsigned short* __restrict__ VT,
                  unsigned short* __restrict__ Ctx) {
    const int qblk = blockIdx.x;
    const int bh   = blockIdx.y;
    const int b    = bh >> 4;
    const int h    = bh & 15;

    const size_t rowbase = (size_t)b * 2048 * 1024 + (size_t)h * 64;
    const unsigned short* Qh = Q + rowbase;                             // [n][d]
    const unsigned short* Kh = K + rowbase;                             // [n][d]
    const unsigned short* Vh = VT + ((size_t)b * 1024 + h * 64) * 2048; // [d][n]

    __shared__ __align__(16) unsigned short Ks[2][32][72];  // [key][d]
    __shared__ __align__(16) unsigned short Vs[2][64][40];  // [d][key]
    __shared__ __align__(16) unsigned short Ps[4][16][32];

    const int tid  = threadIdx.x;
    const int lane = tid & 31;
    const int w    = tid >> 5;
    const int q0   = qblk * 64 + w * 16;

    v16bf qf[2];
#pragma unroll
    for (int s = 0; s < 2; ++s)
        qf[s] = load_a_frag(Qh + (size_t)q0 * 1024 + s * 32, 1024, lane);

    float m_r[8], l_r[8];
#pragma unroll
    for (int j = 0; j < 8; ++j) { m_r[j] = -1e30f; l_r[j] = 0.0f; }
    v8f oacc[4] = {};

    const int ncol = lane & 15;
    const int mofs = (lane & 16) ? 8 : 0;
    const int nblk = 2 * qblk + 2;

    const int rk = tid >> 3, ck = (tid & 7) * 8;   // K map: 16 keys x 64 d / pass
    const int rv = tid >> 2, cv = (tid & 3) * 8;   // V map: 32 d x 32 keys / pass

    auto stage = [&](int blk, int buf) {
#pragma unroll
        for (int it = 0; it < 2; ++it) {
            copy16_g2l(Kh + (size_t)(blk * 32 + rk + it * 16) * 1024 + ck,
                       &Ks[buf][rk + it * 16][ck]);
            copy16_g2l(Vh + (size_t)(rv + it * 32) * 2048 + blk * 32 + cv,
                       &Vs[buf][rv + it * 32][cv]);
        }
    };

    stage(0, 0);
    for (int kb = 0; kb < nblk; ++kb) {
        const int cur = kb & 1, nxt = cur ^ 1;
        const int kbase = kb * 32;
        if (kb + 1 < nblk) { stage(kb + 1, nxt); async_wait_le4(); }
        else               { async_wait_le0(); }
        wg_barrier();

        v8f s[2] = {};
#pragma unroll
        for (int t = 0; t < 2; ++t)
#pragma unroll
            for (int ks = 0; ks < 2; ++ks) {
                v16bf bf = load_b_frag(&Ks[cur][t * 16][ks * 32], 72, lane);
                s[t] = wmma_bf16(qf[ks], bf, s[t]);
            }

        float escale[8];
#pragma unroll
        for (int j = 0; j < 8; ++j) {
            int qrow = q0 + mofs + j;
#pragma unroll
            for (int t = 0; t < 2; ++t) {
                int key = kbase + t * 16 + ncol;
                float v = s[t][j] * 0.125f;
                s[t][j] = (key <= qrow) ? v : -1e30f;
            }
            float rm = fmaxf(s[0][j], s[1][j]);
#pragma unroll
            for (int off = 8; off; off >>= 1)
                rm = fmaxf(rm, __shfl_xor(rm, off, 32));
            float mn  = fmaxf(m_r[j], rm);
            escale[j] = __expf(m_r[j] - mn);
            m_r[j]    = mn;
            float p0 = __expf(s[0][j] - mn);
            float p1 = __expf(s[1][j] - mn);
            s[0][j] = p0; s[1][j] = p1;
            float rsum = p0 + p1;
#pragma unroll
            for (int off = 8; off; off >>= 1)
                rsum += __shfl_xor(rsum, off, 32);
            l_r[j] = l_r[j] * escale[j] + rsum;
        }

        // P: C layout -> A layout via per-wave LDS patch
#pragma unroll
        for (int j = 0; j < 8; ++j) {
            int row = mofs + j;
            Ps[w][row][ncol]      = f32_to_bf16_bits(s[0][j]);
            Ps[w][row][16 + ncol] = f32_to_bf16_bits(s[1][j]);
        }

#pragma unroll
        for (int d = 0; d < 4; ++d)
#pragma unroll
            for (int j = 0; j < 8; ++j)
                oacc[d][j] *= escale[j];

        v16bf pf = load_a_frag(&Ps[w][0][0], 32, lane);
#pragma unroll
        for (int d = 0; d < 4; ++d) {
            v16bf vf = load_b_frag(&Vs[cur][d * 16][0], 40, lane);
            oacc[d] = wmma_bf16(pf, vf, oacc[d]);
        }
        wg_barrier();   // all waves done with buf `cur` before refill
    }

#pragma unroll
    for (int j = 0; j < 8; ++j) {
        int qrow  = q0 + mofs + j;
        float inv = 1.0f / l_r[j];
        unsigned short* dst = Ctx + rowbase + (size_t)qrow * 1024;
#pragma unroll
        for (int d = 0; d < 4; ++d)
            dst[d * 16 + ncol] = f32_to_bf16_bits(oacc[d][j] * inv);
    }
}

// ---------------------------------------------------------------------------
// Output projection: ctx(bf16) @ W_out(bf16, pre-transposed) + b_out -> fp32
// ---------------------------------------------------------------------------
__global__ __launch_bounds__(256)
void k_out_gemm(const unsigned short* __restrict__ Cb,
                const unsigned short* __restrict__ WoT,
                const float* __restrict__ bias,
                float* __restrict__ out) {
    const int M0 = blockIdx.y * 128;
    const int N0 = blockIdx.x * 128;

    __shared__ __align__(16) unsigned short As[2][128][72];
    __shared__ __align__(16) unsigned short Bs[2][128][72];

    const int tid  = threadIdx.x;
    const int lane = tid & 31;
    const int wave = tid >> 5;
    const int wy   = wave >> 2;
    const int wx   = wave & 3;

    v8f acc[4][2] = {};
    gemm_mainloop(Cb, WoT, M0, N0, tid, lane, wy, wx, As, Bs, acc);

    const int ncol = lane & 15;
    const int mofs = (lane & 16) ? 8 : 0;
#pragma unroll
    for (int j = 0; j < 2; ++j) {
        int col = N0 + wx * 32 + j * 16 + ncol;
        float bcol = bias[col];
#pragma unroll
        for (int i = 0; i < 4; ++i) {
            float* prow = out + (size_t)(M0 + wy * 64 + i * 16 + mofs) * 1024 + col;
#pragma unroll
            for (int r = 0; r < 8; ++r)
                prow[(size_t)r * 1024] = acc[i][j][r] + bcol;
        }
    }
}

// ---------------------------------------------------------------------------
// Launch
// ---------------------------------------------------------------------------
extern "C" void kernel_launch(void* const* d_in, const int* in_sizes, int n_in,
                              void* d_out, int out_size, void* d_ws, size_t ws_size,
                              hipStream_t stream) {
    const float* x  = (const float*)d_in[0];
    const float* Wq = (const float*)d_in[1];
    const float* Wk = (const float*)d_in[2];
    const float* Wv = (const float*)d_in[3];
    const float* Wo = (const float*)d_in[4];
    const float* bo = (const float*)d_in[5];
    float* out = (float*)d_out;

    const size_t NX = 8192ull * 1024ull;
    const size_t NW = 1024ull * 1024ull;

    unsigned short* ws  = (unsigned short*)d_ws;
    unsigned short* xb  = ws;
    unsigned short* wqb = xb + NX;     // WqT (pre-transposed bf16)
    unsigned short* wkb = wqb + NW;    // WkT
    unsigned short* wvb = wkb + NW;    // WvT
    unsigned short* wob = wvb + NW;    // WoT
    unsigned short* Qb  = wob + NW;    // [8192][1024] row-major
    unsigned short* Kb  = Qb + NX;     // [8192][1024] row-major
    unsigned short* VTb = Kb + NX;     // [4][1024][2048] transposed
    unsigned short* Cb  = VTb + NX;    // ctx, row-major; ~88 MiB total

    k_f32_to_bf16<<<(int)((NX / 4 + 255) / 256), 256, 0, stream>>>(
        (const float4*)x, (uint2*)xb, (int)(NX / 4));
    k_f32_to_bf16_T<<<dim3(32, 32), 256, 0, stream>>>(Wq, wqb);
    k_f32_to_bf16_T<<<dim3(32, 32), 256, 0, stream>>>(Wk, wkb);
    k_f32_to_bf16_T<<<dim3(32, 32), 256, 0, stream>>>(Wv, wvb);
    k_f32_to_bf16_T<<<dim3(32, 32), 256, 0, stream>>>(Wo, wob);

    k_qkv_gemm<<<dim3(8, 64, 3), 256, 0, stream>>>(xb, wqb, wkb, wvb, Qb, Kb, VTb);
    k_flash_attn<<<dim3(32, 64), 128, 0, stream>>>(Qb, Kb, VTb, Cb);
    k_out_gemm<<<dim3(8, 64), 256, 0, stream>>>(Cb, wob, bo, out);
}
